// EncDecBlock_74577812128016
// MI455X (gfx1250) — matrix operand order repaired and checked by reading, not compiled, not run on hardware
//
#include <hip/hip_runtime.h>
#include <hip/hip_bf16.h>
#include <math.h>

// ---------------------------------------------------------------------------
// MI455X (gfx1250) fused transformer block.
//   B=4, NQ=NKV=1024, QD=1024, KVD=768, H=16, D=64, HID=4096
// All matmuls on v_wmma_f32_16x16x32_f16 (fp32 accumulate).  GEMM B tiles are
// staged into LDS with CDNA5 async copies (global_load_async_to_lds_b128 +
// s_wait_asynccnt), double-buffered across the K loop.  Attention is fused
// flash-style so the only large HBM stream is the unavoidable 256MB alibi.
// ---------------------------------------------------------------------------

typedef _Float16 v16h __attribute__((ext_vector_type(16)));
typedef _Float16 v8h  __attribute__((ext_vector_type(8)));
typedef float    v8f  __attribute__((ext_vector_type(8)));

#define NQ_C   1024
#define NKV_C  1024
#define QD_C   1024
#define KVD_C  768
#define H_C    16
#define D_C    64
#define HID_C  4096
#define MROWS  4096   // B*NQ (== B*NKV here)

__device__ __forceinline__ v8f wmma_f16(v16h a, v16h b, v8f c) {
  // 8 args: (neg_a, A, neg_b, B, c_mod, C, reuse_a, reuse_b)
  return __builtin_amdgcn_wmma_f32_16x16x32_f16(false, a, false, b, (short)0, c,
                                                false, false);
}

// CDNA5 async copy: 16B per lane, global -> LDS, tracked by ASYNCcnt.
// GV addressing: vdst = VGPR holding LDS byte offset, vaddr = 64b global addr.
__device__ __forceinline__ void async_copy_b128(unsigned int lds_off,
                                                const void* gptr) {
  asm volatile("global_load_async_to_lds_b128 %0, %1, off"
               :
               : "v"(lds_off), "v"((unsigned long long)gptr)
               : "memory");
}
__device__ __forceinline__ void wait_async_le1() {
  asm volatile("s_wait_asynccnt 0x1" ::: "memory");
}
__device__ __forceinline__ void wait_async_le0() {
  asm volatile("s_wait_asynccnt 0x0" ::: "memory");
}

// A-fragment (16x32 f16, M x K), row-major source with leading dim `ld`.
// Layout: lanes 0-15 row M=lane hold K 0..15; lanes 16-31 hold
// K 16..31 of the same row.  Two contiguous 16B loads per lane.
__device__ __forceinline__ v16h load_a_frag(const _Float16* __restrict__ base,
                                            int ld) {
  int lane = threadIdx.x & 31;
  int row  = lane & 15;
  int koff = (lane >> 4) * 16;
  const _Float16* p = base + (size_t)row * ld + koff;
  v8h lo = *(const v8h*)(p);
  v8h hi = *(const v8h*)(p + 8);
  v16h r;
#pragma unroll
  for (int i = 0; i < 8; ++i) { r[i] = lo[i]; r[i + 8] = hi[i]; }
  return r;
}

// B-fragment (32x16 f16, K x N) where column n of B is a contiguous row of the
// source (i.e. source = W row-major (N,K), B = W^T).  Lanes 0-15: col n=lane,
// K 0..15; lanes 16-31: col n=lane-16, K 16..31.  One 32B load per lane.
__device__ __forceinline__ v16h load_b_frag(const _Float16* __restrict__ base,
                                            int ld) {
  int lane = threadIdx.x & 31;
  int n    = lane & 15;
  int koff = (lane >> 4) * 16;
  return *(const v16h*)(base + (size_t)n * ld + koff);
}

// ---------------------------------------------------------------------------
// Elementwise f32 -> f16 cast
// ---------------------------------------------------------------------------
__global__ void cast_f32_f16_kernel(const float* __restrict__ in,
                                    _Float16* __restrict__ out, int n) {
  int i = blockIdx.x * blockDim.x + threadIdx.x;
  if (i < n) out[i] = (_Float16)in[i];
}

// ---------------------------------------------------------------------------
// Row LayerNorm (biased var, eps=1e-5) -> fp16 output.  One block per row.
// ---------------------------------------------------------------------------
__global__ __launch_bounds__(256) void ln_kernel(
    const float* __restrict__ in, const float* __restrict__ g,
    const float* __restrict__ bt, _Float16* __restrict__ out, int C) {
  __shared__ float rs[256], rss[256];
  const float* x = in + (size_t)blockIdx.x * C;
  float s = 0.f, ss = 0.f;
  for (int i = threadIdx.x; i < C; i += 256) {
    float v = x[i];
    s += v;
    ss = fmaf(v, v, ss);
  }
  rs[threadIdx.x]  = s;
  rss[threadIdx.x] = ss;
  __syncthreads();
  for (int o = 128; o > 0; o >>= 1) {
    if ((int)threadIdx.x < o) {
      rs[threadIdx.x]  += rs[threadIdx.x + o];
      rss[threadIdx.x] += rss[threadIdx.x + o];
    }
    __syncthreads();
  }
  float mean = rs[0] / C;
  float var  = rss[0] / C - mean * mean;
  float inv  = rsqrtf(var + 1e-5f);
  _Float16* o16 = out + (size_t)blockIdx.x * C;
  for (int i = threadIdx.x; i < C; i += 256)
    o16[i] = (_Float16)((x[i] - mean) * inv * g[i] + bt[i]);
}

// ---------------------------------------------------------------------------
// WMMA GEMM:  C(MxN) = A(MxK, f16 row-major) * W^T  with W (N,K) f16 row-major
// Block = 256 threads = 8 waves; block tile 128(M) x 64(N); wave -> 16x64.
// The 64x32 B tile per K-step is async-copied to LDS (one b128 per lane),
// double buffered: iter i waits asynccnt<=1 (in-order returns => tile i
// resident), computes from buf[i&1], then prefetches tile i+2 into buf[i&1].
// M and N must be multiples of 128 / 64 (all call sites are).
// Epilogues: 0=store f32; 2=+bias+residual f32; 3=+bias,gelu -> f16;
//            4=2*(x+bias) f32 (reference returns m+m).
// ---------------------------------------------------------------------------
template <int EPI>
__global__ __launch_bounds__(256) void gemm_wmma_kernel(
    const _Float16* __restrict__ A, const _Float16* __restrict__ W,
    const float* __restrict__ bias, const float* __restrict__ residual,
    float* __restrict__ outF, _Float16* __restrict__ outH,
    int M, int N, int K) {
  __shared__ _Float16 btile[2][64 * 32];  // [buf][n_row * 32 + k], 4KB each
  int tid  = threadIdx.x;
  int wid  = tid >> 5;
  int lane = tid & 31;
  int m0   = blockIdx.x * 128 + wid * 16;
  int n0   = blockIdx.y * 64;

  // This thread's 16B chunk of the cooperative B-tile copy.
  int brow   = tid >> 2;         // 0..63 : n within tile
  int bchunk = (tid & 3) * 8;    // 0,8,16,24 : k halves within tile row
  unsigned int lds0 =
      (unsigned int)(uintptr_t)&btile[0][brow * 32 + bchunk];
  unsigned int lds1 =
      (unsigned int)(uintptr_t)&btile[1][brow * 32 + bchunk];
  const _Float16* gsrc = W + (size_t)(n0 + brow) * K + bchunk;

  const int T = K / 32;
  async_copy_b128(lds0, gsrc);        // tile 0 -> buf0
  async_copy_b128(lds1, gsrc + 32);   // tile 1 -> buf1

  v8f acc[4] = {};
  const _Float16* arow = A + (size_t)m0 * K;
  for (int i = 0; i < T; ++i) {
    if (i + 1 < T) wait_async_le1(); else wait_async_le0();
    __syncthreads();  // tile i visible to all waves
    v16h a = load_a_frag(arow + i * 32, K);
    const _Float16* bt = &btile[i & 1][0];
#pragma unroll
    for (int t = 0; t < 4; ++t) {
      v16h b = load_b_frag(bt + (size_t)(16 * t) * 32, 32);  // ds_load path
      acc[t] = wmma_f16(a, b, acc[t]);
    }
    __syncthreads();  // all waves done reading buf[i&1]
    if (i + 2 < T)
      async_copy_b128((i & 1) ? lds1 : lds0, gsrc + (size_t)(i + 2) * 32);
  }

  int half = lane >> 4;  // C tile: VGPR r, lane l -> row r+8*(l>>4), col l&15
  int col  = lane & 15;
#pragma unroll
  for (int t = 0; t < 4; ++t) {
#pragma unroll
    for (int r = 0; r < 8; ++r) {
      size_t m = (size_t)(m0 + r + 8 * half);
      size_t n = (size_t)(n0 + 16 * t + col);
      size_t idx = m * (size_t)N + n;
      float v = acc[t][r];
      if (EPI == 0) {
        outF[idx] = v;
      } else if (EPI == 2) {
        outF[idx] = v + bias[n] + residual[idx];
      } else if (EPI == 3) {
        float x = v + bias[n];
        outH[idx] = (_Float16)(0.5f * x * (1.f + erff(x * 0.70710678118f)));
      } else if (EPI == 4) {
        outF[idx] = 2.0f * (v + bias[n]);
      }
    }
  }
}

// ---------------------------------------------------------------------------
// Per-head L2-normalize of q (wave per 64-d vector) -> qhat (b,h,n,d) f16
// ---------------------------------------------------------------------------
__global__ __launch_bounds__(256) void qhead_norm_kernel(
    const float* __restrict__ qh, _Float16* __restrict__ qhat) {
  int gv   = blockIdx.x * 8 + (threadIdx.x >> 5);  // (b*NQ+n)*H + h
  int lane = threadIdx.x & 31;
  int h  = gv & (H_C - 1);
  int bn = gv >> 4;
  const float* src = qh + (size_t)bn * QD_C + h * D_C + lane * 2;
  float v0 = src[0], v1 = src[1];
  float ss = v0 * v0 + v1 * v1;
#pragma unroll
  for (int o = 1; o < 32; o <<= 1) ss += __shfl_xor(ss, o, 32);
  float inv = 1.f / fmaxf(sqrtf(ss), 1e-12f);
  int b = bn >> 10, n = bn & (NQ_C - 1);
  _Float16* dst =
      qhat + ((size_t)(b * H_C + h) * NQ_C + n) * D_C + lane * 2;
  dst[0] = (_Float16)(v0 * inv);
  dst[1] = (_Float16)(v1 * inv);
}

// ---------------------------------------------------------------------------
// Split kv projection rows into normalized khat (b,h,m,d) and transposed
// v (b,h,d,m) -- transpose makes the PV B-fragment a contiguous per-lane load.
// ---------------------------------------------------------------------------
__global__ __launch_bounds__(256) void kv_split_kernel(
    const float* __restrict__ kvF, _Float16* __restrict__ khat,
    _Float16* __restrict__ vT) {
  int gv   = blockIdx.x * 8 + (threadIdx.x >> 5);  // (b*NKV+m)*H + h
  int lane = threadIdx.x & 31;
  int h  = gv & (H_C - 1);
  int bm = gv >> 4;
  int b = bm >> 10, m = bm & (NKV_C - 1);
  // k part: cols [h*64, h*64+64) of the 2*QD-wide projection
  const float* ks = kvF + (size_t)bm * (2 * QD_C) + h * D_C + lane * 2;
  float k0 = ks[0], k1 = ks[1];
  float ss = k0 * k0 + k1 * k1;
#pragma unroll
  for (int o = 1; o < 32; o <<= 1) ss += __shfl_xor(ss, o, 32);
  float inv = 1.f / fmaxf(sqrtf(ss), 1e-12f);
  _Float16* kd =
      khat + ((size_t)(b * H_C + h) * NKV_C + m) * D_C + lane * 2;
  kd[0] = (_Float16)(k0 * inv);
  kd[1] = (_Float16)(k1 * inv);
  // v part: cols [QD + h*64, ...)
  const float* vs = kvF + (size_t)bm * (2 * QD_C) + QD_C + h * D_C + lane * 2;
  size_t vbase = ((size_t)(b * H_C + h) * D_C + lane * 2) * NKV_C + m;
  vT[vbase]          = (_Float16)vs[0];
  vT[vbase + NKV_C]  = (_Float16)vs[1];
}

// ---------------------------------------------------------------------------
// Fused flash attention.  Block = 8 waves; each wave owns one 16-row q tile of
// one (b,h).  Per 32-key step: 4 WMMA for S = qhat * khat^T, alibi + padding
// mask, online softmax (half-wave shuffle reductions match the C-tile row
// mapping), P re-laid out via LDS into an A fragment, 4 WMMA for O += P*V.
// ---------------------------------------------------------------------------
__global__ __launch_bounds__(256) void flash_attn_kernel(
    const _Float16* __restrict__ qhat, const _Float16* __restrict__ khat,
    const _Float16* __restrict__ vT, const float* __restrict__ alibi,
    const unsigned char* __restrict__ pad, const float* __restrict__ logit_scale,
    _Float16* __restrict__ attn_out) {
  __shared__ _Float16 pbuf[8][16 * 32 + 16];  // per-wave P staging (16B-aligned)
  int wid  = threadIdx.x >> 5;
  int lane = threadIdx.x & 31;
  int bh   = blockIdx.x >> 3;             // b*H + h
  int qblk = blockIdx.x & 7;
  int b = bh >> 4, h = bh & (H_C - 1);
  int q0 = qblk * 128 + wid * 16;
  int half = lane >> 4, col = lane & 15;

  float scale = __expf(fminf(logit_scale[h], 4.6051702f));  // log(100)

  // q A-fragments for the full K=64 head dim, loaded once.
  const _Float16* qbase = qhat + ((size_t)bh * NQ_C + q0) * D_C;
  v16h a_lo = load_a_frag(qbase, D_C);        // d 0..31
  v16h a_hi = load_a_frag(qbase + 32, D_C);   // d 32..63

  float mrow[8], lrow[8];
#pragma unroll
  for (int r = 0; r < 8; ++r) { mrow[r] = -1e30f; lrow[r] = 0.f; }
  v8f o[4] = {};

  for (int kt = 0; kt < NKV_C; kt += 32) {
    // ---- S = qhat * khat^T, two 16-key C tiles ----
    v8f s[2];
#pragma unroll
    for (int kh = 0; kh < 2; ++kh) {
      const _Float16* kb = khat + ((size_t)bh * NKV_C + kt + 16 * kh) * D_C;
      v16h b0 = load_b_frag(kb, D_C);       // d 0..31
      v16h b1 = load_b_frag(kb + 32, D_C);  // d 32..63
      v8f sa = {};
      sa = wmma_f16(a_lo, b0, sa);
      sa = wmma_f16(a_hi, b1, sa);
      s[kh] = sa;
    }
    // ---- scale, alibi, padding mask (the 256MB stream) ----
#pragma unroll
    for (int kh = 0; kh < 2; ++kh) {
      int kc = kt + 16 * kh + col;
      bool masked = pad[b * NKV_C + kc] != 0;
      const float* ab =
          alibi + ((size_t)bh * NQ_C + q0 + 8 * half) * NKV_C + kc;
#pragma unroll
      for (int r = 0; r < 8; ++r) {
        float v = s[kh][r] * scale + ab[(size_t)r * NKV_C];
        s[kh][r] = masked ? -1e30f : v;
      }
    }
    // ---- online softmax over the 32 new keys ----
#pragma unroll
    for (int r = 0; r < 8; ++r) {
      float vmax = fmaxf(s[0][r], s[1][r]);
#pragma unroll
      for (int o2 = 1; o2 < 16; o2 <<= 1)
        vmax = fmaxf(vmax, __shfl_xor(vmax, o2, 16));
      float mnew = fmaxf(mrow[r], vmax);
      float f  = __expf(mrow[r] - mnew);
      float p0 = __expf(s[0][r] - mnew);
      float p1 = __expf(s[1][r] - mnew);
      s[0][r] = p0;
      s[1][r] = p1;
      float ps = p0 + p1;
#pragma unroll
      for (int o2 = 1; o2 < 16; o2 <<= 1) ps += __shfl_xor(ps, o2, 16);
      lrow[r] = lrow[r] * f + ps;
      mrow[r] = mnew;
      o[0][r] *= f; o[1][r] *= f; o[2][r] *= f; o[3][r] *= f;
    }
    // ---- stage P (16x32) to LDS row-major, reload as A fragment ----
#pragma unroll
    for (int r = 0; r < 8; ++r) {
      pbuf[wid][(r + 8 * half) * 32 + col]      = (_Float16)s[0][r];
      pbuf[wid][(r + 8 * half) * 32 + 16 + col] = (_Float16)s[1][r];
    }
    __syncthreads();
    v16h ap = load_a_frag(&pbuf[wid][0], 32);
    // ---- O += P * V (V^T layout makes B loads contiguous) ----
#pragma unroll
    for (int t = 0; t < 4; ++t) {
      const _Float16* vb = vT + ((size_t)bh * D_C + 16 * t) * NKV_C + kt;
      v16h bf = load_b_frag(vb, NKV_C);
      o[t] = wmma_f16(ap, bf, o[t]);
    }
    __syncthreads();
  }

  // ---- finalize: O / l, write fp16 (B,NQ,QD) for the out-proj GEMM ----
#pragma unroll
  for (int t = 0; t < 4; ++t) {
#pragma unroll
    for (int r = 0; r < 8; ++r) {
      int q = q0 + r + 8 * half;
      float inv = 1.0f / lrow[r];
      attn_out[(size_t)(b * NQ_C + q) * QD_C + h * D_C + 16 * t + col] =
          (_Float16)(o[t][r] * inv);
    }
  }
}

// ---------------------------------------------------------------------------
// Host: orchestrate the pipeline on `stream` with a static ws bump layout.
// ---------------------------------------------------------------------------
extern "C" void kernel_launch(void* const* d_in, const int* in_sizes, int n_in,
                              void* d_out, int out_size, void* d_ws,
                              size_t ws_size, hipStream_t stream) {
  (void)in_sizes; (void)n_in; (void)out_size; (void)ws_size;
  const float* q     = (const float*)d_in[0];
  const float* kv    = (const float*)d_in[1];
  const unsigned char* pad = (const unsigned char*)d_in[2];  // bool mask
  const float* alibi = (const float*)d_in[3];
  const float* Wq    = (const float*)d_in[4];
  const float* Wkv   = (const float*)d_in[5];
  const float* Wproj = (const float*)d_in[6];
  const float* bproj = (const float*)d_in[7];
  const float* ls    = (const float*)d_in[8];
  const float* ln1g  = (const float*)d_in[9];
  const float* ln1b  = (const float*)d_in[10];
  const float* ln2g  = (const float*)d_in[11];
  const float* ln2b  = (const float*)d_in[12];
  const float* Wfc1  = (const float*)d_in[13];
  const float* bfc1  = (const float*)d_in[14];
  const float* Wfc2  = (const float*)d_in[15];
  const float* bfc2  = (const float*)d_in[16];
  float* out = (float*)d_out;

  char* ws = (char*)d_ws;
  const size_t MB = 1ull << 20;
  // Persistent fp16 weights [0,23MB)
  _Float16* wq_h    = (_Float16*)(ws + 0 * MB);    // 2MB
  _Float16* wkv_h   = (_Float16*)(ws + 2 * MB);    // 3MB
  _Float16* wproj_h = (_Float16*)(ws + 5 * MB);    // 2MB
  _Float16* wfc1_h  = (_Float16*)(ws + 7 * MB);    // 8MB
  _Float16* wfc2_h  = (_Float16*)(ws + 15 * MB);   // 8MB
  // Activations (lifetime-aliased)
  _Float16* qn_h   = (_Float16*)(ws + 23 * MB);    // 8MB  LN1(q) f16
  _Float16* kv_h   = (_Float16*)(ws + 31 * MB);    // 6MB  kv f16
  float*    qhF    = (float*)(ws + 37 * MB);       // 16MB q proj f32
  float*    kvF    = (float*)(ws + 53 * MB);       // 32MB kv proj f32
  _Float16* qhat   = (_Float16*)(ws + 85 * MB);    // 8MB  (b,h,n,d)
  _Float16* khat   = (_Float16*)(ws + 93 * MB);    // 8MB  (b,h,m,d)
  _Float16* vT_h   = (_Float16*)(ws + 101 * MB);   // 8MB  (b,h,d,m)
  _Float16* attn_h = (_Float16*)(ws + 109 * MB);   // 8MB  attn out f16
  float*    xF     = (float*)(ws + 37 * MB);       // 16MB residual (reuse qhF/kvF)
  _Float16* h_h    = (_Float16*)(ws + 53 * MB);    // 8MB  LN2(x) f16
  _Float16* m1_h   = (_Float16*)(ws + 61 * MB);    // 32MB gelu(fc1) f16 (reuse qhat..)

  auto cast = [&](const float* src, _Float16* dst, int n) {
    cast_f32_f16_kernel<<<(n + 255) / 256, 256, 0, stream>>>(src, dst, n);
  };
  cast(Wq, wq_h, QD_C * QD_C);
  cast(Wkv, wkv_h, 2 * QD_C * KVD_C);
  cast(Wproj, wproj_h, QD_C * QD_C);
  cast(Wfc1, wfc1_h, HID_C * QD_C);
  cast(Wfc2, wfc2_h, QD_C * HID_C);
  cast(kv, kv_h, MROWS * KVD_C);

  ln_kernel<<<MROWS, 256, 0, stream>>>(q, ln1g, ln1b, qn_h, QD_C);

  dim3 blk(256);
  // qh = LN1(q) @ Wq^T
  gemm_wmma_kernel<0><<<dim3(MROWS / 128, QD_C / 64), blk, 0, stream>>>(
      qn_h, wq_h, nullptr, nullptr, qhF, nullptr, MROWS, QD_C, QD_C);
  // kvp = kv @ Wkv^T
  gemm_wmma_kernel<0><<<dim3(MROWS / 128, (2 * QD_C) / 64), blk, 0, stream>>>(
      kv_h, wkv_h, nullptr, nullptr, kvF, nullptr, MROWS, 2 * QD_C, KVD_C);

  qhead_norm_kernel<<<(MROWS * H_C) / 8, 256, 0, stream>>>(qhF, qhat);
  kv_split_kernel<<<(MROWS * H_C) / 8, 256, 0, stream>>>(kvF, khat, vT_h);

  // fused cosine attention (4*16 heads * 8 q-blocks = 512 blocks)
  flash_attn_kernel<<<4 * H_C * (NQ_C / 128), 256, 0, stream>>>(
      qhat, khat, vT_h, alibi, pad, ls, attn_h);

  // x = q + attn @ Wproj^T + bproj
  gemm_wmma_kernel<2><<<dim3(MROWS / 128, QD_C / 64), blk, 0, stream>>>(
      attn_h, wproj_h, bproj, q, xF, nullptr, MROWS, QD_C, QD_C);

  ln_kernel<<<MROWS, 256, 0, stream>>>(xF, ln2g, ln2b, h_h, QD_C);

  // m1 = gelu(h @ Wfc1^T + bfc1)
  gemm_wmma_kernel<3><<<dim3(MROWS / 128, HID_C / 64), blk, 0, stream>>>(
      h_h, wfc1_h, bfc1, nullptr, nullptr, m1_h, MROWS, HID_C, QD_C);
  // out = 2 * (m1 @ Wfc2^T + bfc2)
  gemm_wmma_kernel<4><<<dim3(MROWS / 128, QD_C / 64), blk, 0, stream>>>(
      m1_h, wfc2_h, bfc2, nullptr, out, nullptr, MROWS, QD_C, HID_C);
}
